// LSTM_81492709475007
// MI455X (gfx1250) — matrix operand order repaired and checked by reading, not compiled, not run on hardware
//
#include <hip/hip_runtime.h>

// LSTM-like scan, MI455X (gfx1250), wave32 WMMA f16->f32.
// Phase 1: xproj = x @ w_x + b_x  (massively parallel GEMM, f16 WMMA),
//          stored to workspace in a lane-contiguous fragment layout
//          (f16, 128MB total -> resident in the 192MB L2).
// Phase 2: per-batch-row-independent recurrence; 8 workgroups (16 rows each),
//          w_h B-fragments pinned in registers, h double-buffered in LDS.

typedef __attribute__((ext_vector_type(16))) _Float16 v16h;
typedef __attribute__((ext_vector_type(8)))  _Float16 v8h;
typedef __attribute__((ext_vector_type(8)))  float    v8f;

#define BSZ   128
#define TT    2048
#define II    256
#define HH    256
#define LDSP  264          // padded LDS row stride in halfs (bank spread)
#define NTHR  512          // 16 waves of 32

// workspace fragment layout: [t][bblk][wave][lane][r] halfs
// -> per (t,bblk) block = 4096 halfs (8KB); lane's 8 values contiguous (b128)
__device__ inline size_t xp_off(int t, int bblk, int wave, int lane) {
  return (((size_t)t * 8 + bblk) * 16 + wave) * 256 + lane * 8;
}

// ---- B-matrix fragment (32x16 f16, K x N) gathered from global row-major W[K][N]
// element e of lane L:  K = kt*32 + (L<16 ? 0 : 8) + e + (e<8 ? 0 : 8) ,  N = n0 + (L%16)
__device__ inline v16h load_bfrag_global(const float* __restrict__ w, int ldn,
                                         int n0, int kt, int lane) {
  const int kb = kt * 32 + ((lane & 16) ? 8 : 0);
  const int n  = n0 + (lane & 15);
  v16h b;
#pragma unroll
  for (int e = 0; e < 16; ++e)
    b[e] = (_Float16)w[(size_t)(kb + e + (e < 8 ? 0 : 8)) * ldn + n];
  return b;
}

// ---- A-matrix fragment (16x32 f16, M x K) from LDS row-major [16][LDSP]
// lane L: row m = L%16; halves: K in [kt*32 + (L<16?0:8), +8) and [.. +16, +8)
__device__ inline v16h load_afrag_lds(const _Float16* hb, int kt, int lane) {
  const int m  = lane & 15;
  const int k1 = kt * 32 + ((lane & 16) ? 8 : 0);
  const v8h lo = *(const v8h*)(hb + m * LDSP + k1);        // ds_load_b128
  const v8h hi = *(const v8h*)(hb + m * LDSP + k1 + 16);   // ds_load_b128
  v16h a;
#pragma unroll
  for (int e = 0; e < 8; ++e) { a[e] = lo[e]; a[8 + e] = hi[e]; }
  return a;
}

__device__ inline float fast_sigmoid(float z) {
  // 1 / (1 + exp(-z)) via exp2/rcp hardware transcendentals
  return __builtin_amdgcn_rcpf(1.0f + __builtin_amdgcn_exp2f(-1.44269504f * z));
}
__device__ inline float fast_tanh(float z) {
  return 2.0f * fast_sigmoid(2.0f * z) - 1.0f;
}

// =====================  Phase 1: input projection  =====================
// grid = TT blocks of 512 threads. Block t computes all 8 row-blocks of step t,
// reusing the w_x register fragments across them.
__global__ __launch_bounds__(NTHR, 1)
void lstm_xproj_kernel(const float* __restrict__ x, const float* __restrict__ wx,
                       const float* __restrict__ bx, _Float16* __restrict__ xp) {
  __shared__ _Float16 xs[16 * LDSP];
  const int th = threadIdx.x, wave = th >> 5, lane = th & 31;
  const int t  = blockIdx.x;

  v16h bf[8];
#pragma unroll
  for (int kt = 0; kt < 8; ++kt)
    bf[kt] = load_bfrag_global(wx, HH, wave * 16, kt, lane);
  const float bxv = bx[wave * 16 + (lane & 15)];

  for (int bblk = 0; bblk < 8; ++bblk) {
    // stage 16 rows of x (fp32 -> f16) into padded LDS
    {
      const int row = th >> 5;           // 0..15
      const int seg = th & 31;           // floats [seg*8, seg*8+8)
      const float* src = x + ((size_t)(bblk * 16 + row) * TT + t) * II + seg * 8;
      const float4 f0 = *(const float4*)(src);
      const float4 f1 = *(const float4*)(src + 4);
      v8h h8;
      h8[0] = (_Float16)f0.x; h8[1] = (_Float16)f0.y;
      h8[2] = (_Float16)f0.z; h8[3] = (_Float16)f0.w;
      h8[4] = (_Float16)f1.x; h8[5] = (_Float16)f1.y;
      h8[6] = (_Float16)f1.z; h8[7] = (_Float16)f1.w;
      *(v8h*)(xs + row * LDSP + seg * 8) = h8;
    }
    __syncthreads();

    v8f acc = {};
#pragma unroll
    for (int kt = 0; kt < 8; ++kt) {
      const v16h a = load_afrag_lds(xs, kt, lane);
      acc = __builtin_amdgcn_wmma_f32_16x16x32_f16(false, a, false, bf[kt],
                                                   (short)0, acc, false, false);
    }
    // store fragment lane-contiguous: one global_store_b128 per lane
    v8h o8;
#pragma unroll
    for (int r = 0; r < 8; ++r)
      o8[r] = (_Float16)(acc[r] + bxv);
    *(v8h*)(xp + xp_off(t, bblk, wave, lane)) = o8;
    __syncthreads();   // xs reused next bblk
  }
}

// =====================  Phase 2: sequential scan  =====================
// grid = 8 blocks; block handles 16 batch rows through all T steps.
__global__ __launch_bounds__(NTHR, 1)
void lstm_scan_kernel(const _Float16* __restrict__ xp, const float* __restrict__ wh,
                      const float* __restrict__ bh, float* __restrict__ out) {
  __shared__ _Float16 hb[2][16 * LDSP];
  const int th = threadIdx.x, wave = th >> 5, lane = th & 31;
  const int bblk = blockIdx.x;

  for (int i = th; i < 16 * LDSP; i += NTHR) hb[0][i] = (_Float16)0.0f;

  v16h bf[8];
#pragma unroll
  for (int kt = 0; kt < 8; ++kt)
    bf[kt] = load_bfrag_global(wh, HH, wave * 16, kt, lane);
  const float bhv = bh[wave * 16 + (lane & 15)];

  float cst[8], hst[8];
#pragma unroll
  for (int r = 0; r < 8; ++r) { cst[r] = 0.0f; hst[r] = 0.0f; }

  const int mhi  = 8 * (lane >> 4);            // D-row offset
  const int ncol = wave * 16 + (lane & 15);    // D-column (H index)
  const _Float16* xpb = xp + xp_off(0, bblk, wave, lane);

  __syncthreads();

  int cur = 0;
  for (int t = 0; t < TT; ++t) {
    // single coalesced b128 load of this lane's 8 xproj values
    const v8h xr8 = *(const v8h*)(xpb + (size_t)t * (8 * 4096));
    if (t + 1 < TT) {
      const char* pf = (const char*)(xp + ((size_t)(t + 1) * 8 + bblk) * 4096) + th * 16;
      __builtin_prefetch(pf, 0, 1);            // global_prefetch_b8, next step's 8KB
    }

    v8f acc = {};
#pragma unroll
    for (int kt = 0; kt < 8; ++kt) {
      const v16h a = load_afrag_lds(&hb[cur][0], kt, lane);
      acc = __builtin_amdgcn_wmma_f32_16x16x32_f16(false, a, false, bf[kt],
                                                   (short)0, acc, false, false);
    }

    const int nxt = cur ^ 1;
#pragma unroll
    for (int r = 0; r < 8; ++r) {
      const float z = acc[r] + (float)xr8[r] + bhv;
      const float s = fast_sigmoid(z);         // f == i == o
      const float g = fast_tanh(z);
      cst[r] = s * (cst[r] + g);
      hst[r] = s * fast_tanh(cst[r]);
      hb[nxt][(r + mhi) * LDSP + ncol] = (_Float16)hst[r];
    }
    cur = nxt;
    __syncthreads();
  }

  // outputs: h_T then c_T, each [128][256] fp32
#pragma unroll
  for (int r = 0; r < 8; ++r) {
    const int b = bblk * 16 + r + mhi;
    out[(size_t)b * HH + ncol]            = hst[r];
    out[(size_t)BSZ * HH + b * HH + ncol] = cst[r];
  }
}

extern "C" void kernel_launch(void* const* d_in, const int* in_sizes, int n_in,
                              void* d_out, int out_size, void* d_ws, size_t ws_size,
                              hipStream_t stream) {
  const float* x  = (const float*)d_in[0];
  const float* wx = (const float*)d_in[1];
  const float* bx = (const float*)d_in[2];
  const float* wh = (const float*)d_in[3];
  const float* bh = (const float*)d_in[4];
  _Float16* xp = (_Float16*)d_ws;   // T*B*H halfs = 128 MB

  lstm_xproj_kernel<<<TT, NTHR, 0, stream>>>(x, wx, bx, xp);
  lstm_scan_kernel<<<BSZ / 16, NTHR, 0, stream>>>(xp, wh, bh, (float*)d_out);
}
